// GAT_33105607917777
// MI455X (gfx1250) — matrix operand order repaired and checked by reading, not compiled, not run on hardware
//
#include <hip/hip_runtime.h>
#include <hip/hip_bf16.h>

typedef __attribute__((ext_vector_type(16))) _Float16 v16h;
typedef __attribute__((ext_vector_type(8)))  float    v8f;

#define TPB 256
#define HID 128
#define NHEAD 4
#define NGRAPH 128

// ---- monotone float <-> uint encoding for atomicMax on floats ----
__device__ __forceinline__ unsigned fenc(float f) {
  unsigned u = __float_as_uint(f);
  return (u & 0x80000000u) ? ~u : (u | 0x80000000u);
}
__device__ __forceinline__ float fdec(unsigned u) {
  return (u & 0x80000000u) ? __uint_as_float(u ^ 0x80000000u)
                           : __uint_as_float(~u);
}

__global__ void k_fill(float* __restrict__ p, float v, int n) {
  int i = blockIdx.x * blockDim.x + threadIdx.x;
  if (i < n) p[i] = v;
}

// ---------------- WMMA GEMM: Y[M,Nout] = X[M,K] @ W[K,Nout] (+bias) ----------------
// f32 in/out, f16 compute via v_wmma_f32_16x16x32_f16. K % 32 == 0, Nout % 16 == 0.
// A rows are CLAMPED (not predicated) so loads stay unconditional (EXEC all-1s,
// vector b128 loads). Bias handling is a compile-time template; the epilogue
// store guard is wave-uniform and addresses are pointer-stepped.
template <bool HAS_BIAS>
__global__ __launch_bounds__(256) void k_gemm_wmma(
    const float* __restrict__ X, const float* __restrict__ W,
    const float* __restrict__ bias, float* __restrict__ Y,
    int M, int K, int Nout)
{
  const int wave  = threadIdx.x >> 5;
  const int lane  = threadIdx.x & 31;
  const int tilesN = Nout >> 4;
  const int tilesM = (M + 15) >> 4;
  const int tile = blockIdx.x * (blockDim.x >> 5) + wave;
  if (tile >= tilesM * tilesN) return;   // whole-wave uniform exit
  const int tm = tile / tilesN;
  const int tn = tile % tilesN;
  const int half = lane >> 4;           // 0: lanes 0-15, 1: lanes 16-31
  const int lr   = lane & 15;
  int rowA = tm * 16 + lr;
  if (rowA > M - 1) rowA = M - 1;       // clamp: keeps loads in-bounds, no divergence
  const int colB = tn * 16 + lr;
  const float* xr = X + (size_t)rowA * K;

  v8f acc = {};
  for (int kk = 0; kk < K; kk += 32) {
    // ---- A 16x32 f16 layout ----
    // lanes 0-15  (M=lane)   : elems 0..7 = K kk+0..7,  elems 8..15 = K kk+16..23
    // lanes 16-31 (M=lane-16): elems 0..7 = K kk+8..15, elems 8..15 = K kk+24..31
    const int k0 = kk + half * 8;       // multiple of 8 floats -> 32B aligned
    const float4 a0 = *(const float4*)(xr + k0);
    const float4 a1 = *(const float4*)(xr + k0 + 4);
    const float4 a2 = *(const float4*)(xr + k0 + 16);
    const float4 a3 = *(const float4*)(xr + k0 + 20);
    v16h a;
    a[0]  = (_Float16)a0.x; a[1]  = (_Float16)a0.y;
    a[2]  = (_Float16)a0.z; a[3]  = (_Float16)a0.w;
    a[4]  = (_Float16)a1.x; a[5]  = (_Float16)a1.y;
    a[6]  = (_Float16)a1.z; a[7]  = (_Float16)a1.w;
    a[8]  = (_Float16)a2.x; a[9]  = (_Float16)a2.y;
    a[10] = (_Float16)a2.z; a[11] = (_Float16)a2.w;
    a[12] = (_Float16)a3.x; a[13] = (_Float16)a3.y;
    a[14] = (_Float16)a3.z; a[15] = (_Float16)a3.w;

    // ---- B 32x16 f16 layout: lane = column; lanes 0-15 K=kk+0..7,16..23,
    //      lanes 16-31 K=kk+8..15,24..31. Batch the 16 independent strided loads
    //      (coalesced across lanes) before converting so waits are staggered.
    const int kb = kk + half * 8;
    const float* wp = W + (size_t)kb * Nout + colB;
    float bt[16];
#pragma unroll
    for (int j = 0; j < 16; ++j) bt[j] = wp[(size_t)(j + (j >= 8 ? 8 : 0)) * Nout];
    v16h b;
#pragma unroll
    for (int j = 0; j < 16; ++j) b[j] = (_Float16)bt[j];

    acc = __builtin_amdgcn_wmma_f32_16x16x32_f16(
        /*neg_a=*/false, a, /*neg_b=*/false, b,
        /*c_mod=*/(short)0, acc, /*reuse_a=*/false, /*reuse_b=*/false);
  }

  // ---- epilogue ----
  // C/D layout: VGPR r -> M = r (lanes 0-15), M = r+8 (lanes 16-31); N = lane&15
  float bv = 0.f;
  if (HAS_BIAS) bv = bias[colB];            // one load, same column for all rows
  const int row0 = tm * 16 + half * 8;
  float* yp = Y + (size_t)row0 * Nout + colB;
  const bool fullTile = (tm * 16 + 16) <= M;  // wave-uniform
  if (fullTile) {
#pragma unroll
    for (int r = 0; r < 8; ++r) {
      *yp = HAS_BIAS ? acc[r] + bv : acc[r];
      yp += Nout;
    }
  } else {
#pragma unroll
    for (int r = 0; r < 8; ++r) {
      if (row0 + r < M) *yp = HAS_BIAS ? acc[r] + bv : acc[r];
      yp += Nout;
    }
  }
}

// ---------------- GAT attention coefficients: e_s/e_d [N,H] ----------------
__global__ void k_att_coef(const float* __restrict__ h,
                           const float* __restrict__ a_src,
                           const float* __restrict__ a_dst,
                           float* __restrict__ es, float* __restrict__ ed, int N) {
  int i = blockIdx.x * blockDim.x + threadIdx.x;
  if (i >= N * NHEAD) return;
  int n = i >> 2, hh = i & 3;
  const float* hp = h + (size_t)n * HID + hh * 32;
  const float* as = a_src + hh * 32;
  const float* ad = a_dst + hh * 32;
  float s = 0.f, d = 0.f;
#pragma unroll 8
  for (int c = 0; c < 32; ++c) { float v = hp[c]; s += v * as[c]; d += v * ad[c]; }
  es[i] = s; ed[i] = d;
}

__device__ __forceinline__ void edge_sd(const int* __restrict__ ei, int e, int E,
                                        int& s, int& d) {
  if (e < E) { s = ei[e]; d = ei[E + e]; } else { s = e - E; d = e - E; }
}

// segment-max of leaky_relu(e_s[src]+e_d[dst]) over incoming edges of dst
__global__ void k_alpha_max(const float* __restrict__ es, const float* __restrict__ ed,
                            const int* __restrict__ ei, unsigned* __restrict__ mEnc,
                            int E, int Et) {
  int idx = blockIdx.x * blockDim.x + threadIdx.x;
  if (idx >= Et * NHEAD) return;
  int e = idx >> 2, hh = idx & 3, s, d;
  edge_sd(ei, e, E, s, d);
  float al = es[s * NHEAD + hh] + ed[d * NHEAD + hh];
  al = al > 0.f ? al : 0.2f * al;
  atomicMax(&mEnc[d * NHEAD + hh], fenc(al));
}

__global__ void k_alpha_sum(const float* __restrict__ es, const float* __restrict__ ed,
                            const int* __restrict__ ei, const unsigned* __restrict__ mEnc,
                            float* __restrict__ z, int E, int Et) {
  int idx = blockIdx.x * blockDim.x + threadIdx.x;
  if (idx >= Et * NHEAD) return;
  int e = idx >> 2, hh = idx & 3, s, d;
  edge_sd(ei, e, E, s, d);
  float al = es[s * NHEAD + hh] + ed[d * NHEAD + hh];
  al = al > 0.f ? al : 0.2f * al;
  atomicAdd(&z[d * NHEAD + hh], __expf(al - fdec(mEnc[d * NHEAD + hh])));
}

// out[dst,:] += alpha_e[head] * h[src,:]   (128 lanes per edge; atomics land in L2)
__global__ void k_scatter(const float* __restrict__ h, const float* __restrict__ es,
                          const float* __restrict__ ed, const unsigned* __restrict__ mEnc,
                          const float* __restrict__ z, const int* __restrict__ ei,
                          float* __restrict__ out, int E, int Et) {
  long long idx = (long long)blockIdx.x * blockDim.x + threadIdx.x;
  if (idx >= (long long)Et * HID) return;
  int e = (int)(idx >> 7), f = (int)(idx & 127), hh = f >> 5, s, d;
  edge_sd(ei, e, E, s, d);
  float al = es[s * NHEAD + hh] + ed[d * NHEAD + hh];
  al = al > 0.f ? al : 0.2f * al;
  float w = __expf(al - fdec(mEnc[d * NHEAD + hh])) / (z[d * NHEAD + hh] + 1e-16f);
  atomicAdd(&out[(size_t)d * HID + f], w * h[(size_t)s * HID + f]);
}

__global__ void k_add_bias(float* __restrict__ y, const float* __restrict__ b, int total) {
  int i = blockIdx.x * blockDim.x + threadIdx.x;
  if (i < total) y[i] += b[i & (HID - 1)];
}

// ---------------- GraphNorm ----------------
__global__ void k_gn_acc(const float* __restrict__ x, const int* __restrict__ bat,
                         float* __restrict__ gsum, float* __restrict__ cnt, int N) {
  long long idx = (long long)blockIdx.x * blockDim.x + threadIdx.x;
  if (idx >= (long long)N * HID) return;
  int n = (int)(idx >> 7), f = (int)(idx & 127);
  int g = bat[n];
  atomicAdd(&gsum[g * HID + f], x[idx]);
  if (f == 0) atomicAdd(&cnt[g], 1.0f);
}
__global__ void k_gn_mean(float* __restrict__ gsum, const float* __restrict__ cnt, int n) {
  int i = blockIdx.x * blockDim.x + threadIdx.x;
  if (i >= n) return;
  float c = cnt[i >> 7];
  gsum[i] = c > 0.f ? gsum[i] / c : 0.f;
}
__global__ void k_gn_pass2(float* __restrict__ x, const int* __restrict__ bat,
                           const float* __restrict__ mean, const float* __restrict__ ms,
                           float* __restrict__ gvar, int N) {
  long long idx = (long long)blockIdx.x * blockDim.x + threadIdx.x;
  if (idx >= (long long)N * HID) return;
  int n = (int)(idx >> 7), f = (int)(idx & 127);
  int g = bat[n];
  float v = x[idx] - mean[g * HID + f] * ms[f];
  x[idx] = v;
  atomicAdd(&gvar[g * HID + f], v * v);
}
__global__ void k_gn_fin(float* __restrict__ gvar, const float* __restrict__ cnt, int n) {
  int i = blockIdx.x * blockDim.x + threadIdx.x;
  if (i >= n) return;
  float c = cnt[i >> 7];
  float var = c > 0.f ? gvar[i] / c : 0.f;
  gvar[i] = rsqrtf(var + 1e-5f);
}
__global__ void k_gn_apply(float* __restrict__ x, const int* __restrict__ bat,
                           const float* __restrict__ istd, const float* __restrict__ w,
                           const float* __restrict__ b, int N) {
  long long idx = (long long)blockIdx.x * blockDim.x + threadIdx.x;
  if (idx >= (long long)N * HID) return;
  int n = (int)(idx >> 7), f = (int)(idx & 127);
  int g = bat[n];
  float v = w[f] * x[idx] * istd[g * HID + f] + b[f];
  x[idx] = v > 0.f ? v : 0.f;   // fused ReLU
}

// ---------------- BN(eval, mean=0, var=1) + ReLU ----------------
__global__ void k_bn_relu(float* __restrict__ y, const float* __restrict__ g,
                          const float* __restrict__ b, int F, int total) {
  int i = blockIdx.x * blockDim.x + threadIdx.x;
  if (i >= total) return;
  const float invs = 0.9999950000374997f;  // 1/sqrt(1+1e-5)
  int f = i % F;
  float v = g[f] * y[i] * invs + b[f];
  y[i] = v > 0.f ? v : 0.f;
}

// ---------------- gate second linear: [N,64] @ [64,1] + b ----------------
__global__ void k_gate2(const float* __restrict__ gb, const float* __restrict__ w,
                        const float* __restrict__ b, float* __restrict__ g, int N) {
  int n = blockIdx.x * blockDim.x + threadIdx.x;
  if (n >= N) return;
  const float* r = gb + (size_t)n * 64;
  float s = 0.f;
#pragma unroll 8
  for (int f = 0; f < 64; ++f) s += r[f] * w[f];
  g[n] = s + b[0];
}

__global__ void k_nmax(const float* __restrict__ g, const int* __restrict__ bat,
                       unsigned* __restrict__ gmE, int N) {
  int n = blockIdx.x * blockDim.x + threadIdx.x;
  if (n < N) atomicMax(&gmE[bat[n]], fenc(g[n]));
}
__global__ void k_nsum(const float* __restrict__ g, const int* __restrict__ bat,
                       const unsigned* __restrict__ gmE, float* __restrict__ gz, int N) {
  int n = blockIdx.x * blockDim.x + threadIdx.x;
  if (n >= N) return;
  int b = bat[n];
  atomicAdd(&gz[b], __expf(g[n] - fdec(gmE[b])));
}
__global__ void k_pool(const float* __restrict__ x, const float* __restrict__ g,
                       const int* __restrict__ bat, const unsigned* __restrict__ gmE,
                       const float* __restrict__ gz, float* __restrict__ pooled, int N) {
  long long idx = (long long)blockIdx.x * blockDim.x + threadIdx.x;
  if (idx >= (long long)N * HID) return;
  int n = (int)(idx >> 7), f = (int)(idx & 127);
  int b = bat[n];
  float gate = __expf(g[n] - fdec(gmE[b])) / (gz[b] + 1e-16f);
  atomicAdd(&pooled[b * HID + f], gate * x[idx]);
}

// ---------------- final linear [128,256] @ [256,2] ----------------
__global__ void k_final(const float* __restrict__ h, const float* __restrict__ w,
                        const float* __restrict__ b, float* __restrict__ out) {
  int i = threadIdx.x;  // 256 = 128 graphs * 2 outputs
  if (i >= NGRAPH * 2) return;
  int bg = i >> 1, o = i & 1;
  const float* r = h + (size_t)bg * 256;
  float s = 0.f;
  for (int f = 0; f < 256; ++f) s += r[f] * w[f * 2 + o];
  out[i] = s + b[o];
}

static inline int cdiv_ll(long long a, long long b) { return (int)((a + b - 1) / b); }

extern "C" void kernel_launch(void* const* d_in, const int* in_sizes, int n_in,
                              void* d_out, int out_size, void* d_ws, size_t ws_size,
                              hipStream_t stream)
{
  const float* x   = (const float*)d_in[0];
  const int*   ei  = (const int*)d_in[2];
  const int*   bat = (const int*)d_in[3];
  const float *W1 = (const float*)d_in[4],  *as1 = (const float*)d_in[5];
  const float *ad1= (const float*)d_in[6],  *b1  = (const float*)d_in[7];
  const float *gn1w=(const float*)d_in[8],  *gn1b=(const float*)d_in[9],  *gn1ms=(const float*)d_in[10];
  const float *W2 = (const float*)d_in[11], *as2 = (const float*)d_in[12];
  const float *ad2= (const float*)d_in[13], *b2  = (const float*)d_in[14];
  const float *gn2w=(const float*)d_in[15], *gn2b=(const float*)d_in[16], *gn2ms=(const float*)d_in[17];
  const float *g1W=(const float*)d_in[18],  *g1b =(const float*)d_in[19];
  const float *g1g=(const float*)d_in[20],  *g1bb=(const float*)d_in[21];
  const float *g2W=(const float*)d_in[22],  *g2b =(const float*)d_in[23];
  const float *l1W=(const float*)d_in[24],  *l1b =(const float*)d_in[25];
  const float *bn1g=(const float*)d_in[26], *bn1b=(const float*)d_in[27];
  const float *l2W=(const float*)d_in[28],  *l2b =(const float*)d_in[29];
  const float *bn2g=(const float*)d_in[30], *bn2b=(const float*)d_in[31];
  const float *loW=(const float*)d_in[32],  *lob =(const float*)d_in[33];
  float* out = (float*)d_out;

  const int N  = in_sizes[0] / HID;
  const int E  = in_sizes[2] / 2;
  const int Et = E + N;   // + self loops

  // ---- carve workspace ----
  float* ws = (float*)d_ws;
  size_t off = 0;
  auto alloc = [&](size_t n) { float* p = ws + off; off += n; return p; };
  float*    bufH  = alloc((size_t)N * HID);
  float*    bufO  = alloc((size_t)N * HID);
  float*    es    = alloc((size_t)N * NHEAD);
  float*    ed    = alloc((size_t)N * NHEAD);
  float*    zbuf  = alloc((size_t)N * NHEAD);
  unsigned* mEnc  = (unsigned*)alloc((size_t)N * NHEAD);
  float*    gbuf  = alloc((size_t)N * 64);
  float*    gnode = alloc((size_t)N);
  float*    gsum  = alloc(NGRAPH * HID);
  float*    gvar  = alloc(NGRAPH * HID);
  float*    cnt   = alloc(NGRAPH);
  unsigned* gmE   = (unsigned*)alloc(NGRAPH);
  float*    gz    = alloc(NGRAPH);
  float*    pooled= alloc(NGRAPH * HID);
  float*    m1    = alloc(NGRAPH * 256);
  float*    m2    = alloc(NGRAPH * 256);
  (void)ws_size; (void)n_in; (void)out_size;

  const float* Ws[2]  = {W1, W2};
  const float* As[2]  = {as1, as2};
  const float* Ad[2]  = {ad1, ad2};
  const float* Bs[2]  = {b1, b2};
  const float* GNw[2] = {gn1w, gn2w};
  const float* GNb[2] = {gn1b, gn2b};
  const float* GNm[2] = {gn1ms, gn2ms};

  const float* xin = x;
  const long long NH = (long long)N * HID;

  for (int L = 0; L < 2; ++L) {
    // h = x @ W  (WMMA)
    int tiles = ((N + 15) / 16) * (HID / 16);
    k_gemm_wmma<false><<<cdiv_ll(tiles, 8), 256, 0, stream>>>(xin, Ws[L], nullptr, bufH, N, HID, HID);
    k_att_coef<<<cdiv_ll((long long)N * NHEAD, TPB), TPB, 0, stream>>>(bufH, As[L], Ad[L], es, ed, N);
    // reset accumulators (every call: graph replays reuse ws)
    k_fill<<<cdiv_ll(NH, TPB), TPB, 0, stream>>>(bufO, 0.f, N * HID);
    k_fill<<<cdiv_ll((long long)N * NHEAD, TPB), TPB, 0, stream>>>((float*)mEnc, 0.f, N * NHEAD);
    k_fill<<<cdiv_ll((long long)N * NHEAD, TPB), TPB, 0, stream>>>(zbuf, 0.f, N * NHEAD);
    // segment softmax + scatter
    k_alpha_max<<<cdiv_ll((long long)Et * NHEAD, TPB), TPB, 0, stream>>>(es, ed, ei, mEnc, E, Et);
    k_alpha_sum<<<cdiv_ll((long long)Et * NHEAD, TPB), TPB, 0, stream>>>(es, ed, ei, mEnc, zbuf, E, Et);
    k_scatter<<<cdiv_ll((long long)Et * HID, TPB), TPB, 0, stream>>>(bufH, es, ed, mEnc, zbuf, ei, bufO, E, Et);
    k_add_bias<<<cdiv_ll(NH, TPB), TPB, 0, stream>>>(bufO, Bs[L], N * HID);
    // GraphNorm + ReLU (in place in bufO)
    k_fill<<<cdiv_ll(NGRAPH * HID, TPB), TPB, 0, stream>>>(gsum, 0.f, NGRAPH * HID);
    k_fill<<<cdiv_ll(NGRAPH * HID, TPB), TPB, 0, stream>>>(gvar, 0.f, NGRAPH * HID);
    k_fill<<<1, NGRAPH, 0, stream>>>(cnt, 0.f, NGRAPH);
    k_gn_acc<<<cdiv_ll(NH, TPB), TPB, 0, stream>>>(bufO, bat, gsum, cnt, N);
    k_gn_mean<<<cdiv_ll(NGRAPH * HID, TPB), TPB, 0, stream>>>(gsum, cnt, NGRAPH * HID);
    k_gn_pass2<<<cdiv_ll(NH, TPB), TPB, 0, stream>>>(bufO, bat, gsum, GNm[L], gvar, N);
    k_gn_fin<<<cdiv_ll(NGRAPH * HID, TPB), TPB, 0, stream>>>(gvar, cnt, NGRAPH * HID);
    k_gn_apply<<<cdiv_ll(NH, TPB), TPB, 0, stream>>>(bufO, bat, gvar, GNw[L], GNb[L], N);
    xin = bufO;
  }

  // ---- GlobalAttention pooling ----
  {
    int tiles = ((N + 15) / 16) * (64 / 16);
    k_gemm_wmma<true><<<cdiv_ll(tiles, 8), 256, 0, stream>>>(bufO, g1W, g1b, gbuf, N, HID, 64);
  }
  k_bn_relu<<<cdiv_ll((long long)N * 64, TPB), TPB, 0, stream>>>(gbuf, g1g, g1bb, 64, N * 64);
  k_gate2<<<cdiv_ll(N, TPB), TPB, 0, stream>>>(gbuf, g2W, g2b, gnode, N);
  k_fill<<<1, NGRAPH, 0, stream>>>((float*)gmE, 0.f, NGRAPH);
  k_fill<<<1, NGRAPH, 0, stream>>>(gz, 0.f, NGRAPH);
  k_fill<<<cdiv_ll(NGRAPH * HID, TPB), TPB, 0, stream>>>(pooled, 0.f, NGRAPH * HID);
  k_nmax<<<cdiv_ll(N, TPB), TPB, 0, stream>>>(gnode, bat, gmE, N);
  k_nsum<<<cdiv_ll(N, TPB), TPB, 0, stream>>>(gnode, bat, gmE, gz, N);
  k_pool<<<cdiv_ll(NH, TPB), TPB, 0, stream>>>(bufO, gnode, bat, gmE, gz, pooled, N);

  // ---- MLP head ----
  {
    int tiles = (NGRAPH / 16) * (256 / 16);
    k_gemm_wmma<true><<<cdiv_ll(tiles, 8), 256, 0, stream>>>(pooled, l1W, l1b, m1, NGRAPH, HID, 256);
    k_bn_relu<<<cdiv_ll(NGRAPH * 256, TPB), TPB, 0, stream>>>(m1, bn1g, bn1b, 256, NGRAPH * 256);
    k_gemm_wmma<true><<<cdiv_ll(tiles, 8), 256, 0, stream>>>(m1, l2W, l2b, m2, NGRAPH, 256, 256);
    k_bn_relu<<<cdiv_ll(NGRAPH * 256, TPB), TPB, 0, stream>>>(m2, bn2g, bn2b, 256, NGRAPH * 256);
  }
  k_final<<<1, 256, 0, stream>>>(m2, loW, lob, out);
}
